// AttentionBlock_24970939859079
// MI455X (gfx1250) — matrix operand order repaired and checked by reading, not compiled, not run on hardware
//
#include <hip/hip_runtime.h>

typedef __attribute__((ext_vector_type(16))) _Float16 v16h;
typedef __attribute__((ext_vector_type(8)))  float    v8f;
typedef _Float16 f16;

static constexpr int kB  = 2;
static constexpr int kC  = 512;   // channels
static constexpr int kS  = 4096;  // 64*64 spatial
static constexpr int kNH = 8;
static constexpr int kDH = 64;
static constexpr int kIn = 512;   // inner = NH*DH

// ---------------------------------------------------------------------------
// WMMA + fragment helpers (CDNA5 v_wmma_f32_16x16x32_f16, layouts ISA 7.12.2)
// ---------------------------------------------------------------------------
static __device__ __forceinline__ v8f wmma32(v16h a, v16h b, v8f c) {
  return __builtin_amdgcn_wmma_f32_16x16x32_f16(false, a, false, b, (short)0, c,
                                                false, false);
}

union Frag16 {
  v16h h;
  uint4 q[2];
};

// A fragment: 16x32 f16 tile of row-major A (lda), origin (m0,k0).
// lane&15 = row; lanes 0-15 K {0..7,16..23}, lanes 16-31 K {8..15,24..31}.
static __device__ __forceinline__ v16h load_a_frag(const f16* A, int lda,
                                                   int m0, int k0) {
  const int lane = threadIdx.x & 31;
  const f16* p = A + (size_t)(m0 + (lane & 15)) * lda + k0 + (lane >> 4) * 8;
  Frag16 r;
  r.q[0] = *(const uint4*)(p);       // K hk .. hk+7
  r.q[1] = *(const uint4*)(p + 16);  // K 16+hk .. 16+hk+7
  return r.h;
}

// B fragment where logical B[k][n] is stored transposed as Bt[n][k] (ldt):
// lane = col (lane&15); lanes 0-15 K {0..7,16..23}, lanes 16-31 K {8..15,24..31}.
static __device__ __forceinline__ v16h load_bt_frag(const f16* Bt, int ldt,
                                                    int k0, int n0) {
  const int lane = threadIdx.x & 31;
  const f16* p = Bt + (size_t)(n0 + (lane & 15)) * ldt + k0 + (lane >> 4) * 8;
  Frag16 r;
  r.q[0] = *(const uint4*)(p);
  r.q[1] = *(const uint4*)(p + 16);
  return r.h;
}

static __device__ __forceinline__ float rowmax16(float x) {
#pragma unroll
  for (int m = 1; m < 16; m <<= 1) x = fmaxf(x, __shfl_xor(x, m, 32));
  return x;
}
static __device__ __forceinline__ float rowsum16(float x) {
#pragma unroll
  for (int m = 1; m < 16; m <<= 1) x += __shfl_xor(x, m, 32);
  return x;
}

// CDNA5 async memory->LDS copy (ASYNCcnt tracked), ISA §10 / §15.18.3.
// lds: generic shared pointer; LDS offset is addr[31:0] (LDS aperture rule).
static __device__ __forceinline__ void async_ld_b128(void* lds, const f16* g) {
  uint32_t l = (uint32_t)(uintptr_t)lds;
  asm volatile("global_load_async_to_lds_b128 %0, %1, off" ::"v"(l), "v"(g)
               : "memory");
}
static __device__ __forceinline__ void wait_async0() {
  asm volatile("s_wait_asynccnt 0" ::: "memory");
}

// ---------------------------------------------------------------------------
// Stage 0: layout conversion to f16
// ---------------------------------------------------------------------------
// x [B, C, S] f32  ->  Xh [B, S, C] f16
__global__ void cvt_input_kernel(const float* __restrict__ in,
                                 f16* __restrict__ out) {
  int idx = blockIdx.x * blockDim.x + threadIdx.x;  // over B*C*S
  int b = idx / (kC * kS);
  int r = idx - b * (kC * kS);
  int c = r / kS;
  int s = r - c * kS;
  out[(size_t)b * kS * kC + (size_t)s * kC + c] = (f16)in[idx];
}

// Plain f32 -> f16 copy (native weight layout is already Bt[n][k]).
__global__ void cvt_w_f16_kernel(const float* __restrict__ in,
                                 f16* __restrict__ out) {
  int idx = blockIdx.x * blockDim.x + threadIdx.x;  // over 512*512
  out[idx] = (f16)in[idx];
}

// ---------------------------------------------------------------------------
// Async tile staging helpers (128-thread blocks)
// ---------------------------------------------------------------------------
static __device__ __forceinline__ void stage_gemm_tiles(const f16* A,
                                                        const f16* Bt,
                                                        f16* AsB, f16* BtsB,
                                                        int m0, int n0, int ks,
                                                        int lda, int ldb,
                                                        int t) {
#pragma unroll
  for (int i = 0; i < 2; ++i) {
    int idx = t + i * 128;
    int r = idx >> 2, p = idx & 3;
    async_ld_b128(&AsB[r * 32 + p * 8],
                  &A[(size_t)(m0 + r) * lda + ks + p * 8]);
    async_ld_b128(&BtsB[r * 32 + p * 8],
                  &Bt[(size_t)(n0 + r) * ldb + ks + p * 8]);
  }
}

static __device__ __forceinline__ void stage_attn_tiles(const f16* Kb,
                                                        const f16* Vb,
                                                        f16* KsB, f16* VtsB,
                                                        int c0, int t) {
#pragma unroll
  for (int i = 0; i < 4; ++i) {
    int idx = t + i * 128;
    int r = idx >> 3, p = idx & 7;
    async_ld_b128(&KsB[r * 64 + p * 8], &Kb[(size_t)(c0 + r) * kDH + p * 8]);
    async_ld_b128(&VtsB[r * 64 + p * 8], &Vb[(size_t)r * kS + c0 + p * 8]);
  }
}

// ---------------------------------------------------------------------------
// Stage 1: QKV projection GEMM  [S x C] * Wt^T (+bias)
// grid: (S/64, Inner/64, B*3), block 128 (4 waves). Async double-buffered
// LDS tiles. Q,K written [h][s][d]; V written TRANSPOSED [h][d][s].
// ---------------------------------------------------------------------------
__global__ void qkv_gemm_kernel(const f16* __restrict__ Xh,
                                const f16* __restrict__ Chx,
                                const f16* __restrict__ Wqh,
                                const f16* __restrict__ Wkh,
                                const f16* __restrict__ Wvh,
                                const float* __restrict__ bq,
                                const float* __restrict__ bk,
                                const float* __restrict__ bv,
                                f16* __restrict__ Qh, f16* __restrict__ Kh,
                                f16* __restrict__ Vt) {
  __shared__ __align__(16) f16 As[2][64 * 32];   // A tile, row-major
  __shared__ __align__(16) f16 Bts[2][64 * 32];  // Bt tile: [n][k]

  const int z = blockIdx.z;
  const int b = z / 3, j = z - 3 * b;
  const f16* A = (j == 0 ? Xh : Chx) + (size_t)b * kS * kC;
  const f16* W = (j == 0 ? Wqh : (j == 1 ? Wkh : Wvh));  // [Inner][C] = Bt
  const float* bias = (j == 0 ? bq : (j == 1 ? bk : bv));

  const int t = threadIdx.x;
  const int wave = t >> 5;
  const int m0 = blockIdx.x * 64, n0 = blockIdx.y * 64;
  constexpr int kSteps = kC / 32;

  stage_gemm_tiles(A, W, As[0], Bts[0], m0, n0, 0, kC, kC, t);
  wait_async0();
  __syncthreads();

  v8f acc[4] = {};
  for (int kk = 0; kk < kSteps; ++kk) {
    const int cur = kk & 1;
    if (kk + 1 < kSteps)  // prefetch next tile while computing this one
      stage_gemm_tiles(A, W, As[cur ^ 1], Bts[cur ^ 1], m0, n0, (kk + 1) * 32,
                       kC, kC, t);
    v16h a = load_a_frag(As[cur], 32, wave * 16, 0);
#pragma unroll
    for (int nt = 0; nt < 4; ++nt)
      acc[nt] = wmma32(a, load_bt_frag(Bts[cur], 32, 0, nt * 16), acc[nt]);
    wait_async0();
    __syncthreads();
  }

  // epilogue: bias + scatter
  const int lane15 = t & 15, half = (t & 31) >> 4;
  const int head = blockIdx.y;  // 64-aligned N tile == one head
#pragma unroll
  for (int nt = 0; nt < 4; ++nt) {
    int d = nt * 16 + lane15;
    float bv_ = bias[head * 64 + d];
#pragma unroll
    for (int v = 0; v < 8; ++v) {
      int sr = m0 + wave * 16 + v + 8 * half;
      f16 val = (f16)(acc[nt][v] + bv_);
      if (j == 2) {  // V transposed: [b][h][d][s]
        f16* Out = Vt + (size_t)b * kNH * kS * kDH;
        Out[((size_t)head * kDH + d) * kS + sr] = val;
      } else {  // Q/K: [b][h][s][d]
        f16* Out = (j == 0 ? Qh : Kh) + (size_t)b * kNH * kS * kDH;
        Out[((size_t)head * kS + sr) * kDH + d] = val;
      }
    }
  }
}

// ---------------------------------------------------------------------------
// Stage 2: flash attention. grid (S/64, NH, B), block 128 (4 waves).
// Wave w owns rows [rb*64 + w*16, +16). K/V tiles double-buffered in LDS,
// staged with CDNA5 async global->LDS b128 copies overlapped with WMMA.
// ---------------------------------------------------------------------------
__global__ void attention_kernel(const f16* __restrict__ Qh,
                                 const f16* __restrict__ Kh,
                                 const f16* __restrict__ Vt,
                                 f16* __restrict__ O) {
  __shared__ __align__(16) f16 Ks[2][64 * 64];   // [col][d]  (Bt for Q*K^T)
  __shared__ __align__(16) f16 Vts[2][64 * 64];  // [d][col]  (Bt for P*V)
  __shared__ __align__(16) f16 Ps[4 * 16 * 64];

  const int t = threadIdx.x, wave = t >> 5;
  const int lane15 = t & 15, half = (t & 31) >> 4;
  const int rb = blockIdx.x, h = blockIdx.y, b = blockIdx.z;
  const size_t hoff = (size_t)(b * kNH + h) * kS * kDH;
  const f16* Qb = Qh + hoff;  // [s][d]
  const f16* Kb = Kh + hoff;  // [s][d]
  const f16* Vb = Vt + hoff;  // [d][s]

  // Q strip fragments, pre-scaled by 1/sqrt(64) = 0.125 (exact in f16)
  v16h aq0 = load_a_frag(Qb, kDH, rb * 64 + wave * 16, 0);
  v16h aq1 = load_a_frag(Qb, kDH, rb * 64 + wave * 16, 32);
#pragma unroll
  for (int i = 0; i < 16; ++i) {
    aq0[i] = (f16)((float)aq0[i] * 0.125f);
    aq1[i] = (f16)((float)aq1[i] * 0.125f);
  }

  v8f acc[4] = {};
  float mrun[8], lrun[8];
#pragma unroll
  for (int v = 0; v < 8; ++v) { mrun[v] = -1e30f; lrun[v] = 0.f; }

  f16* Pw = Ps + wave * 16 * 64;  // wave-private P strip [16 x 64]
  constexpr int kTiles = kS / 64;

  stage_attn_tiles(Kb, Vb, Ks[0], Vts[0], 0, t);
  wait_async0();
  __syncthreads();

  for (int ct = 0; ct < kTiles; ++ct) {
    const int cur = ct & 1;
    if (ct + 1 < kTiles)  // async-prefetch next K/V tile during compute
      stage_attn_tiles(Kb, Vb, Ks[cur ^ 1], Vts[cur ^ 1], (ct + 1) * 64, t);

    // S = Q*K^T
    v8f sf[4] = {};
#pragma unroll
    for (int jt = 0; jt < 4; ++jt) {
      sf[jt] = wmma32(aq0, load_bt_frag(Ks[cur], 64, 0, jt * 16), sf[jt]);
      sf[jt] = wmma32(aq1, load_bt_frag(Ks[cur], 64, 32, jt * 16), sf[jt]);
    }

    // online softmax (row r = v + 8*half lives across a 16-lane half)
#pragma unroll
    for (int v = 0; v < 8; ++v) {
      float mloc = fmaxf(fmaxf(sf[0][v], sf[1][v]), fmaxf(sf[2][v], sf[3][v]));
      mloc = rowmax16(mloc);
      float mnew = fmaxf(mrun[v], mloc);
      float corr = __expf(mrun[v] - mnew);
      mrun[v] = mnew;
      float rs = 0.f;
#pragma unroll
      for (int jt = 0; jt < 4; ++jt) {
        float pv = __expf(sf[jt][v] - mnew);
        sf[jt][v] = pv;
        rs += pv;
      }
      lrun[v] = lrun[v] * corr + rowsum16(rs);
#pragma unroll
      for (int dt = 0; dt < 4; ++dt) acc[dt][v] *= corr;
    }

    // P -> wave-private LDS (C-frag layout -> row-major), reload as A-frags
#pragma unroll
    for (int jt = 0; jt < 4; ++jt)
#pragma unroll
      for (int v = 0; v < 8; ++v)
        Pw[(v + 8 * half) * 64 + jt * 16 + lane15] = (f16)sf[jt][v];
    asm volatile("s_wait_dscnt 0" ::: "memory");  // cross-lane LDS WAR
    v16h ap0 = load_a_frag(Pw, 64, 0, 0);
    v16h ap1 = load_a_frag(Pw, 64, 0, 32);

    // acc += P * V   (B[k=col][n=d] = Vts[d][col] -> contiguous Bt reads)
#pragma unroll
    for (int dt = 0; dt < 4; ++dt) {
      acc[dt] = wmma32(ap0, load_bt_frag(Vts[cur], 64, 0, dt * 16), acc[dt]);
      acc[dt] = wmma32(ap1, load_bt_frag(Vts[cur], 64, 32, dt * 16), acc[dt]);
    }

    wait_async0();    // next tile landed in LDS
    __syncthreads();  // all waves done with cur + all asyncs complete
  }

  // epilogue: normalize, write O [b][s][inner] f16
  f16* Ob = O + (size_t)b * kS * kIn;
#pragma unroll
  for (int v = 0; v < 8; ++v) {
    float inv = 1.0f / lrun[v];
    int sr = rb * 64 + wave * 16 + v + 8 * half;
#pragma unroll
    for (int dt = 0; dt < 4; ++dt)
      Ob[(size_t)sr * kIn + h * 64 + dt * 16 + lane15] =
          (f16)(acc[dt][v] * inv);
  }
}

// ---------------------------------------------------------------------------
// Stage 3: output projection  O[S x Inner] * Wo^T + bo -> [B,C,H,W] f32
// Wo [C][Inner] is already Bt[n=c][k=e]. grid (S/64, C/64, B), block 128.
// ---------------------------------------------------------------------------
__global__ void oproj_gemm_kernel(const f16* __restrict__ O,
                                  const f16* __restrict__ Woh,
                                  const float* __restrict__ bo,
                                  float* __restrict__ out) {
  __shared__ __align__(16) f16 As[2][64 * 32];
  __shared__ __align__(16) f16 Bts[2][64 * 32];

  const int t = threadIdx.x, wave = t >> 5;
  const int b = blockIdx.z;
  const int m0 = blockIdx.x * 64, n0 = blockIdx.y * 64;
  const f16* A = O + (size_t)b * kS * kIn;
  constexpr int kSteps = kIn / 32;

  stage_gemm_tiles(A, Woh, As[0], Bts[0], m0, n0, 0, kIn, kIn, t);
  wait_async0();
  __syncthreads();

  v8f acc[4] = {};
  for (int kk = 0; kk < kSteps; ++kk) {
    const int cur = kk & 1;
    if (kk + 1 < kSteps)
      stage_gemm_tiles(A, Woh, As[cur ^ 1], Bts[cur ^ 1], m0, n0,
                       (kk + 1) * 32, kIn, kIn, t);
    v16h a = load_a_frag(As[cur], 32, wave * 16, 0);
#pragma unroll
    for (int nt = 0; nt < 4; ++nt)
      acc[nt] = wmma32(a, load_bt_frag(Bts[cur], 32, 0, nt * 16), acc[nt]);
    wait_async0();
    __syncthreads();
  }

  const int lane15 = t & 15, half = (t & 31) >> 4;
#pragma unroll
  for (int nt = 0; nt < 4; ++nt) {
    int cc = n0 + nt * 16 + lane15;
    float bv_ = bo[cc];
#pragma unroll
    for (int v = 0; v < 8; ++v) {
      int sr = m0 + wave * 16 + v + 8 * half;
      out[(size_t)b * kC * kS + (size_t)cc * kS + sr] = acc[nt][v] + bv_;
    }
  }
}

// ---------------------------------------------------------------------------
// Host launcher
// ---------------------------------------------------------------------------
extern "C" void kernel_launch(void* const* d_in, const int* in_sizes, int n_in,
                              void* d_out, int out_size, void* d_ws,
                              size_t ws_size, hipStream_t stream) {
  const float* x   = (const float*)d_in[0];
  const float* ctx = (const float*)d_in[1];
  const float* Wq  = (const float*)d_in[2];
  const float* bq  = (const float*)d_in[3];
  const float* Wk  = (const float*)d_in[4];
  const float* bk  = (const float*)d_in[5];
  const float* Wv  = (const float*)d_in[6];
  const float* bv  = (const float*)d_in[7];
  const float* Wo  = (const float*)d_in[8];
  const float* bo  = (const float*)d_in[9];
  float* out = (float*)d_out;

  char* ws = (char*)d_ws;
  size_t off = 0;
  auto alloc = [&](size_t bytes) {
    void* p = ws + off;
    off = (off + bytes + 255) & ~(size_t)255;
    return p;
  };
  f16* Xh  = (f16*)alloc((size_t)kB * kS * kC * 2);
  f16* Chx = (f16*)alloc((size_t)kB * kS * kC * 2);
  f16* Wqh = (f16*)alloc((size_t)kC * kIn * 2);
  f16* Wkh = (f16*)alloc((size_t)kC * kIn * 2);
  f16* Wvh = (f16*)alloc((size_t)kC * kIn * 2);
  f16* Woh = (f16*)alloc((size_t)kIn * kC * 2);
  f16* Qh  = (f16*)alloc((size_t)kB * kNH * kS * kDH * 2);
  f16* Kh  = (f16*)alloc((size_t)kB * kNH * kS * kDH * 2);
  f16* Vt  = (f16*)alloc((size_t)kB * kNH * kS * kDH * 2);
  f16* O   = (f16*)alloc((size_t)kB * kS * kIn * 2);
  (void)ws_size; (void)n_in; (void)in_sizes; (void)out_size;

  // Stage 0: conversions
  cvt_input_kernel<<<(kB * kC * kS) / 256, 256, 0, stream>>>(x, Xh);
  cvt_input_kernel<<<(kB * kC * kS) / 256, 256, 0, stream>>>(ctx, Chx);
  cvt_w_f16_kernel<<<(kC * kIn) / 256, 256, 0, stream>>>(Wq, Wqh);
  cvt_w_f16_kernel<<<(kC * kIn) / 256, 256, 0, stream>>>(Wk, Wkh);
  cvt_w_f16_kernel<<<(kC * kIn) / 256, 256, 0, stream>>>(Wv, Wvh);
  cvt_w_f16_kernel<<<(kC * kIn) / 256, 256, 0, stream>>>(Wo, Woh);

  // Stage 1: QKV projections (V written transposed per head)
  qkv_gemm_kernel<<<dim3(kS / 64, kIn / 64, kB * 3), 128, 0, stream>>>(
      Xh, Chx, Wqh, Wkh, Wvh, bq, bk, bv, Qh, Kh, Vt);

  // Stage 2: flash attention (async double-buffered K/V staging)
  attention_kernel<<<dim3(kS / 64, kNH, kB), 128, 0, stream>>>(Qh, Kh, Vt, O);

  // Stage 3: output projection
  oproj_gemm_kernel<<<dim3(kS / 64, kC / 64, kB), 128, 0, stream>>>(O, Woh, bo,
                                                                    out);
}
